// Model_76759655514562
// MI455X (gfx1250) — matrix operand order repaired and checked by reading, not compiled, not run on hardware
//
#include <hip/hip_runtime.h>
#include <math.h>

#define B_   16
#define S_   512
#define CIN_ 64
#define D_   512
#define H_   8
#define E_   64
#define DFF_ 2048
#define LDEC_ 512
#define M_   (B_*S_)     // 8192 rows for both encoder and decoder streams
#define BH_  (B_*H_)     // 128 attention batches
#define PH_  128
#define KIM_ 192         // im2col K = CIN*3

typedef _Float16 h8   __attribute__((ext_vector_type(8)));
typedef _Float16 v16h __attribute__((ext_vector_type(16)));
typedef float    v8f  __attribute__((ext_vector_type(8)));

__device__ __forceinline__ float gelu_exact(float x) {
  return 0.5f * x * (1.0f + erff(x * 0.70710678118654752440f));
}

// ---------------- f32 -> f16 convert ----------------
__global__ void k_cvt16(const float* __restrict__ src, _Float16* __restrict__ dst, int n) {
  int i = blockIdx.x * blockDim.x + threadIdx.x;
  if (i < n) dst[i] = (_Float16)src[i];
}

// ---------------- WMMA NT GEMM, register-blocked 32x64 per wave ----------------
// C[M,N] = A[M,K] * B[N,K]^T (+bias[N]) (+resid) ; optional exact GELU ; f32/f16 out.
// Batched via blockIdx.y: per-batch offsets (bh/Hh)*s?b + (bh%Hh)*s?h (element units).
// Each wave: 2 M-frags x 4 N-frags of 16x16 -> 8 accumulators, K stepped by 32.
// Per K-step: 12 b128 loads feed 8 v_wmma_f32_16x16x32_f16 (A reused 4x, B reused 2x).
__global__ void k_gemm_wmma(
    const _Float16* __restrict__ A, const _Float16* __restrict__ Bw,
    const float* __restrict__ bias, const float* __restrict__ resid,
    float* __restrict__ C32, _Float16* __restrict__ C16,
    int M, int N, int K, int lda, int ldb, int ldc,
    long sAb, long sAh, long sBb, long sBh, long sCb, long sCh,
    int Hh, int fuse)
{
  const int bh = blockIdx.y;
  const int bb = bh / Hh, hh = bh % Hh;
  const _Float16* Ab = A + bb * sAb + hh * sAh;
  const _Float16* Bb = Bw + bb * sBb + hh * sBh;
  const long coff = bb * sCb + hh * sCh;

  const int tilesN4 = N >> 6;                       // N / 64
  const int wave = blockIdx.x * 8 + ((int)threadIdx.x >> 5);
  const int tM2 = wave / tilesN4;                   // 32-row macro tile
  const int tN4 = wave % tilesN4;                   // 64-col macro tile
  if (tM2 * 32 >= M) return;                        // wave-uniform: EXEC stays all-ones

  const int lane = threadIdx.x & 31;
  const int hs = lane >> 4;                         // half-wave select
  const int r  = lane & 15;

  // A lane slices: row = tM2*32 + mi*16 + r, halfs {k0+hs*8..+8} and {k0+16+hs*8..+8}
  const _Float16* Ap0 = Ab + (long)(tM2 * 32 + r) * lda + hs * 8;
  const _Float16* Ap1 = Ap0 + (long)16 * lda;
  // B lane slices: col = tN4*64 + ni*16 + r, halfs {k0+hs*8..+8} and {k0+16+hs*8..+8}
  const _Float16* Bp0 = Bb + (long)(tN4 * 64 + r) * ldb + hs * 8;
  const _Float16* Bp1 = Bp0 + (long)16 * ldb;
  const _Float16* Bp2 = Bp0 + (long)32 * ldb;
  const _Float16* Bp3 = Bp0 + (long)48 * ldb;

  v8f acc[2][4];
#pragma unroll
  for (int mi = 0; mi < 2; ++mi)
#pragma unroll
    for (int ni = 0; ni < 4; ++ni) acc[mi][ni] = (v8f){};

  for (int k0 = 0; k0 < K; k0 += 32) {
    __builtin_prefetch(Ap0 + k0 + 128, 0, 1);
    __builtin_prefetch(Ap1 + k0 + 128, 0, 1);
    __builtin_prefetch(Bp0 + k0 + 128, 0, 1);
    __builtin_prefetch(Bp1 + k0 + 128, 0, 1);
    __builtin_prefetch(Bp2 + k0 + 128, 0, 1);
    __builtin_prefetch(Bp3 + k0 + 128, 0, 1);

    v16h av[2], bv[4];
    {
      h8 x0 = *(const h8*)(Ap0 + k0);
      h8 x1 = *(const h8*)(Ap0 + k0 + 16);
      h8 y0 = *(const h8*)(Ap1 + k0);
      h8 y1 = *(const h8*)(Ap1 + k0 + 16);
#pragma unroll
      for (int i = 0; i < 8; ++i) {
        av[0][i] = x0[i]; av[0][i + 8] = x1[i];
        av[1][i] = y0[i]; av[1][i + 8] = y1[i];
      }
    }
    {
      h8 b00 = *(const h8*)(Bp0 + k0);
      h8 b01 = *(const h8*)(Bp0 + k0 + 16);
      h8 b10 = *(const h8*)(Bp1 + k0);
      h8 b11 = *(const h8*)(Bp1 + k0 + 16);
      h8 b20 = *(const h8*)(Bp2 + k0);
      h8 b21 = *(const h8*)(Bp2 + k0 + 16);
      h8 b30 = *(const h8*)(Bp3 + k0);
      h8 b31 = *(const h8*)(Bp3 + k0 + 16);
#pragma unroll
      for (int i = 0; i < 8; ++i) {
        bv[0][i] = b00[i]; bv[0][i + 8] = b01[i];
        bv[1][i] = b10[i]; bv[1][i + 8] = b11[i];
        bv[2][i] = b20[i]; bv[2][i + 8] = b21[i];
        bv[3][i] = b30[i]; bv[3][i + 8] = b31[i];
      }
    }
#pragma unroll
    for (int mi = 0; mi < 2; ++mi)
#pragma unroll
      for (int ni = 0; ni < 4; ++ni)
        acc[mi][ni] = __builtin_amdgcn_wmma_f32_16x16x32_f16(
            false, av[mi], false, bv[ni], (short)0, acc[mi][ni], false, false);
  }

  // C layout per 16x16 frag: lane r -> col, VGPR v -> row hs*8+v
#pragma unroll
  for (int ni = 0; ni < 4; ++ni) {
    const int col = tN4 * 64 + ni * 16 + r;
    const float bval = bias ? bias[col] : 0.0f;
#pragma unroll
    for (int mi = 0; mi < 2; ++mi) {
#pragma unroll
      for (int v = 0; v < 8; ++v) {
        long row = (long)tM2 * 32 + mi * 16 + hs * 8 + v;
        long idx = coff + row * (long)ldc + col;
        float x = acc[mi][ni][v] + bval;
        if (resid) x += resid[idx];
        if (fuse == 2) x = gelu_exact(x);
        if (C32) C32[idx] = x;
        if (C16) C16[idx] = (_Float16)x;
      }
    }
  }
}

// ---------------- series stats: mean/std over S per (b,c) ----------------
__global__ void k_stats(const float* __restrict__ x, float* __restrict__ mean,
                        float* __restrict__ stdv) {
  int row = blockIdx.x;                  // B*CIN
  int b = row / CIN_, c = row % CIN_;
  int lane = threadIdx.x;
  float s = 0.f, s2 = 0.f;
  for (int t = lane; t < S_; t += 32) {
    float v = x[((size_t)b * S_ + t) * CIN_ + c];
    s += v; s2 += v * v;
  }
  for (int o = 16; o; o >>= 1) { s += __shfl_xor(s, o, 32); s2 += __shfl_xor(s2, o, 32); }
  if (lane == 0) {
    float m = s / (float)S_;
    float var = s2 / (float)S_ - m * m;
    mean[row] = m;
    stdv[row] = sqrtf(var + 1e-5f);
  }
}

__global__ void k_norm_in(const float* __restrict__ x, const float* __restrict__ mean,
                          const float* __restrict__ stdv, float* __restrict__ xe, int n) {
  int i = blockIdx.x * blockDim.x + threadIdx.x;
  if (i >= n) return;
  int c = i % CIN_;
  int b = i / (S_ * CIN_);
  xe[i] = (x[i] - mean[b * CIN_ + c]) / stdv[b * CIN_ + c];
}

__global__ void k_make_xdec(const float* __restrict__ xe, float* __restrict__ xd, int n) {
  int i = blockIdx.x * blockDim.x + threadIdx.x;
  if (i >= n) return;
  int c = i % CIN_;
  int s = (i / CIN_) % LDEC_;
  int b = i / (LDEC_ * CIN_);
  xd[i] = (s < 256) ? xe[((size_t)b * S_ + 256 + s) * CIN_ + c] : 0.0f;
}

// ---------------- projector conv: y[b,e] = sum_{s,k} x[b,s,circ(e+k-1)] * cw[s,k] ----------------
__global__ void k_pconv(const float* __restrict__ x, const float* __restrict__ cw,
                        float* __restrict__ y) {
  int row = blockIdx.x;                  // B*CIN
  int b = row / CIN_, e = row % CIN_;
  int lane = threadIdx.x;
  float acc = 0.f;
  for (int s = lane; s < S_; s += 32) {
#pragma unroll
    for (int k = 0; k < 3; ++k) {
      int ei = (e + k - 1 + CIN_) % CIN_;
      acc += x[((size_t)b * S_ + s) * CIN_ + ei] * cw[s * 3 + k];
    }
  }
  for (int o = 16; o; o >>= 1) acc += __shfl_xor(acc, o, 32);
  if (lane == 0) y[row] = acc;
}

// ---------------- tiny FC: out[b,n] = act( [in1|in2] . W[n,:] + bias ) ----------------
__global__ void k_fc(const float* __restrict__ in1, int len1,
                     const float* __restrict__ in2, int len2,
                     const float* __restrict__ W, const float* __restrict__ bias,
                     float* __restrict__ out, int N, int act) {
  int idx = blockIdx.x * blockDim.x + threadIdx.x;
  if (idx >= B_ * N) return;
  int b = idx / N, n = idx % N;
  const float* w = W + (size_t)n * (len1 + len2);
  float a = bias ? bias[n] : 0.f;
  for (int k = 0; k < len1; ++k) a += in1[b * len1 + k] * w[k];
  for (int k = 0; k < len2; ++k) a += in2[b * len2 + k] * w[len1 + k];
  if (act == 1) a = fmaxf(a, 0.f);
  else if (act == 2) a = expf(a);
  out[idx] = a;
}

// ---------------- circular-conv im2col (K = CIN*3, order c*3+k matches W layout) ----------------
__global__ void k_im2col(const float* __restrict__ src, _Float16* __restrict__ dst, int n) {
  int i = blockIdx.x * blockDim.x + threadIdx.x;
  if (i >= n) return;
  int rkc = i % KIM_;
  int m = i / KIM_;
  int c = rkc / 3, k = rkc % 3;
  int b = m / S_, s = m % S_;
  int sp = (s + k - 1 + S_) % S_;
  dst[i] = (_Float16)src[((size_t)b * S_ + sp) * CIN_ + c];
}

// ---------------- add pos-embed + temporal embed, emit f32 + f16 ----------------
__global__ void k_embed_add(float* __restrict__ emb, _Float16* __restrict__ emb16,
                            const float* __restrict__ mark, const float* __restrict__ tW, int n) {
  int i = blockIdx.x * blockDim.x + threadIdx.x;
  if (i >= n) return;
  int d = i % D_;
  int m = i / D_;
  int s = m % S_;
  int dd = d & ~1;
  float div = expf((float)dd * (-9.210340371976184f / (float)D_));  // ln(10000)
  float ang = (float)s * div;
  float pe = (d & 1) ? cosf(ang) : sinf(ang);
  float v = emb[i] + pe;
#pragma unroll
  for (int j = 0; j < 4; ++j) v += mark[(size_t)m * 4 + j] * tW[d * 4 + j];
  emb[i] = v;
  emb16[i] = (_Float16)v;
}

// ---------------- layernorm over D=512, one wave per row, f32 + optional f16 out ----------------
__global__ void k_layernorm(const float* __restrict__ X, const float* __restrict__ g,
                            const float* __restrict__ bta, float* __restrict__ Y,
                            _Float16* __restrict__ Y16) {
  int row = blockIdx.x;
  int lane = threadIdx.x;
  const float* x = X + (size_t)row * D_;
  float v[16];
  float s = 0.f, s2 = 0.f;
#pragma unroll
  for (int i = 0; i < 16; ++i) {
    v[i] = x[lane + i * 32];
    s += v[i]; s2 += v[i] * v[i];
  }
  for (int o = 16; o; o >>= 1) { s += __shfl_xor(s, o, 32); s2 += __shfl_xor(s2, o, 32); }
  float m = s / (float)D_;
  float var = s2 / (float)D_ - m * m;
  float rs = rsqrtf(var + 1e-5f);
#pragma unroll
  for (int i = 0; i < 16; ++i) {
    int d = lane + i * 32;
    float o = (v[i] - m) * rs * g[d] + bta[d];
    Y[(size_t)row * D_ + d] = o;
    if (Y16) Y16[(size_t)row * D_ + d] = (_Float16)o;
  }
}

// ---------------- de-stationary softmax: (s*tau + delta) [causal] * scale, rows of 512 ----------------
__global__ void k_softmax(const float* __restrict__ Sc, _Float16* __restrict__ P,
                          const float* __restrict__ tau, const float* __restrict__ delta,
                          int Lq, int causal, float scale) {
  int row = blockIdx.x;                  // BH*Lq
  int bh = row / Lq, l = row % Lq;
  int b = bh / H_;
  const float* src = Sc + (size_t)row * S_;
  int lane = threadIdx.x;
  float t = tau[b];
  float vals[16];
  float vmax = -__builtin_inff();
#pragma unroll
  for (int i = 0; i < 16; ++i) {
    int s = lane + i * 32;
    float v = src[s] * t;
    if (delta) v += delta[b * S_ + s];
    if (causal && s > l) v = -__builtin_inff();
    v *= scale;
    vals[i] = v;
    vmax = fmaxf(vmax, v);
  }
  for (int o = 16; o; o >>= 1) vmax = fmaxf(vmax, __shfl_xor(vmax, o, 32));
  float sum = 0.f;
#pragma unroll
  for (int i = 0; i < 16; ++i) { vals[i] = expf(vals[i] - vmax); sum += vals[i]; }
  for (int o = 16; o; o >>= 1) sum += __shfl_xor(sum, o, 32);
  float inv = 1.0f / sum;
#pragma unroll
  for (int i = 0; i < 16; ++i) {
    int s = lane + i * 32;
    P[(size_t)row * S_ + s] = (_Float16)(vals[i] * inv);
  }
}

// ---------------- V transpose: vt[b,h,e,s] = src[(b*S+s)*ld + off + h*E + e] ----------------
__global__ void k_vtrans(const _Float16* __restrict__ src, _Float16* __restrict__ dst,
                         int ld, int off, int n) {
  int i = blockIdx.x * blockDim.x + threadIdx.x;
  if (i >= n) return;
  int s = i % S_;
  int e = (i / S_) % E_;
  int bh = i / (S_ * E_);
  int b = bh / H_, h = bh % H_;
  dst[i] = src[((size_t)b * S_ + s) * ld + off + h * E_ + e];
}

// ---------------- de-normalize output ----------------
__global__ void k_denorm(const float* __restrict__ pin, const float* __restrict__ mean,
                         const float* __restrict__ stdv, float* __restrict__ out, int n) {
  int i = blockIdx.x * blockDim.x + threadIdx.x;
  if (i >= n) return;
  int c = i % CIN_;
  int b = i / (LDEC_ * CIN_);
  out[i] = pin[i] * stdv[b * CIN_ + c] + mean[b * CIN_ + c];
}

extern "C" void kernel_launch(void* const* d_in, const int* in_sizes, int n_in,
                              void* d_out, int out_size, void* d_ws, size_t ws_size,
                              hipStream_t stream) {
  (void)in_sizes; (void)n_in; (void)out_size; (void)ws_size;

  const float* x_enc      = (const float*)d_in[0];
  const float* x_mark_enc = (const float*)d_in[1];
  const float* x_mark_dec = (const float*)d_in[3];
  const float* enc_conv_w = (const float*)d_in[4];
  const float* dec_conv_w = (const float*)d_in[5];
  const float* temp_W_enc = (const float*)d_in[6];
  const float* temp_W_dec = (const float*)d_in[7];
  const float* enc_Wqkv   = (const float*)d_in[8];
  const float* enc_bqkv   = (const float*)d_in[9];
  const float* enc_Wo     = (const float*)d_in[10];
  const float* enc_bo     = (const float*)d_in[11];
  const float* enc_Wf1    = (const float*)d_in[12];
  const float* enc_bf1    = (const float*)d_in[13];
  const float* enc_Wf2    = (const float*)d_in[14];
  const float* enc_bf2    = (const float*)d_in[15];
  const float* enc_ln_g   = (const float*)d_in[16];
  const float* enc_ln_b   = (const float*)d_in[17];
  const float* enc_norm_g = (const float*)d_in[18];
  const float* enc_norm_b = (const float*)d_in[19];
  const float* dec_Wqkv   = (const float*)d_in[20];
  const float* dec_bqkv   = (const float*)d_in[21];
  const float* dec_Wo     = (const float*)d_in[22];
  const float* dec_bo     = (const float*)d_in[23];
  const float* dec_Wf1    = (const float*)d_in[24];
  const float* dec_bf1    = (const float*)d_in[25];
  const float* dec_Wf2    = (const float*)d_in[26];
  const float* dec_bf2    = (const float*)d_in[27];
  const float* dec_ln_g   = (const float*)d_in[28];
  const float* dec_ln_b   = (const float*)d_in[29];
  const float* dec_norm_g = (const float*)d_in[30];
  const float* dec_norm_b = (const float*)d_in[31];
  const float* proj_W     = (const float*)d_in[32];
  const float* proj_b     = (const float*)d_in[33];
  const float* tau_conv   = (const float*)d_in[34];
  const float* tau_W0     = (const float*)d_in[35];
  const float* tau_b0     = (const float*)d_in[36];
  const float* tau_W1     = (const float*)d_in[37];
  const float* tau_b1     = (const float*)d_in[38];
  const float* tau_Wout   = (const float*)d_in[39];
  const float* delta_conv = (const float*)d_in[40];
  const float* delta_W0   = (const float*)d_in[41];
  const float* delta_b0   = (const float*)d_in[42];
  const float* delta_W1   = (const float*)d_in[43];
  const float* delta_b1   = (const float*)d_in[44];
  const float* delta_Wout = (const float*)d_in[45];

  char* ws = (char*)d_ws;
  size_t off = 0;
  auto alloc = [&](size_t bytes) -> char* {
    char* p = ws + off;
    off += (bytes + 255) & ~(size_t)255;
    return p;
  };

  // activations / scratch
  float* mean   = (float*)alloc((size_t)B_ * CIN_ * 4);
  float* stdv   = (float*)alloc((size_t)B_ * CIN_ * 4);
  float* xe     = (float*)alloc((size_t)M_ * CIN_ * 4);
  float* xdn    = (float*)alloc((size_t)M_ * CIN_ * 4);
  float* py     = (float*)alloc((size_t)B_ * CIN_ * 4);
  float* ph0    = (float*)alloc((size_t)B_ * PH_ * 4);
  float* ph1    = (float*)alloc((size_t)B_ * PH_ * 4);
  float* tauv   = (float*)alloc((size_t)B_ * 4);
  float* deltav = (float*)alloc((size_t)B_ * S_ * 4);
  _Float16* im2c = (_Float16*)alloc((size_t)M_ * KIM_ * 2);
  float* cur     = (float*)alloc((size_t)M_ * D_ * 4);
  _Float16* cur16 = (_Float16*)alloc((size_t)M_ * D_ * 2);
  float* encF    = (float*)alloc((size_t)M_ * D_ * 4);
  _Float16* encF16 = (_Float16*)alloc((size_t)M_ * D_ * 2);
  float* dcur    = (float*)alloc((size_t)M_ * D_ * 4);
  _Float16* dcur16 = (_Float16*)alloc((size_t)M_ * D_ * 2);
  float* dF      = (float*)alloc((size_t)M_ * D_ * 4);
  _Float16* dF16 = (_Float16*)alloc((size_t)M_ * D_ * 2);
  float* t1      = (float*)alloc((size_t)M_ * D_ * 4);
  float* t1n     = (float*)alloc((size_t)M_ * D_ * 4);
  _Float16* t1n16 = (_Float16*)alloc((size_t)M_ * D_ * 2);
  float* t2      = (float*)alloc((size_t)M_ * D_ * 4);
  float* t2n     = (float*)alloc((size_t)M_ * D_ * 4);
  _Float16* t2n16 = (_Float16*)alloc((size_t)M_ * D_ * 2);
  _Float16* qkv16 = (_Float16*)alloc((size_t)M_ * 3 * D_ * 2);
  _Float16* qg16  = (_Float16*)alloc((size_t)M_ * D_ * 2);
  _Float16* kv16  = (_Float16*)alloc((size_t)M_ * 2 * D_ * 2);
  _Float16* vt16  = (_Float16*)alloc((size_t)BH_ * E_ * S_ * 2);
  float* scores   = (float*)alloc((size_t)BH_ * S_ * S_ * 4);
  _Float16* probs16 = (_Float16*)alloc((size_t)BH_ * S_ * S_ * 2);
  _Float16* ctx16 = (_Float16*)alloc((size_t)M_ * D_ * 2);
  _Float16* ffh16 = (_Float16*)alloc((size_t)M_ * DFF_ * 2);
  float* pout     = (float*)alloc((size_t)M_ * CIN_ * 4);

  // f16 weights
  _Float16* w_encQKV = (_Float16*)alloc((size_t)2 * 3 * D_ * D_ * 2);
  _Float16* w_encWo  = (_Float16*)alloc((size_t)2 * D_ * D_ * 2);
  _Float16* w_encF1  = (_Float16*)alloc((size_t)2 * DFF_ * D_ * 2);
  _Float16* w_encF2  = (_Float16*)alloc((size_t)2 * D_ * DFF_ * 2);
  _Float16* w_decQKV = (_Float16*)alloc((size_t)2 * 3 * D_ * D_ * 2);
  _Float16* w_decWo  = (_Float16*)alloc((size_t)2 * D_ * D_ * 2);
  _Float16* w_decF1  = (_Float16*)alloc((size_t)DFF_ * D_ * 2);
  _Float16* w_decF2  = (_Float16*)alloc((size_t)D_ * DFF_ * 2);
  _Float16* w_encC   = (_Float16*)alloc((size_t)D_ * KIM_ * 2);
  _Float16* w_decC   = (_Float16*)alloc((size_t)D_ * KIM_ * 2);
  _Float16* w_proj   = (_Float16*)alloc((size_t)CIN_ * D_ * 2);

  auto cvt = [&](const float* s, _Float16* d, int n) {
    k_cvt16<<<(n + 255) / 256, 256, 0, stream>>>(s, d, n);
  };
  auto gemmB = [&](const _Float16* A, const _Float16* Bw, const float* bias,
                   const float* resid, float* C32, _Float16* C16,
                   int M, int N, int K, int lda, int ldb, int ldc,
                   long sAb, long sAh, long sBb, long sBh, long sCb, long sCh,
                   int Hh, int batch, int fuse) {
    int waves = (M / 32) * (N / 64);
    dim3 grid((waves + 7) / 8, batch);
    k_gemm_wmma<<<grid, 256, 0, stream>>>(A, Bw, bias, resid, C32, C16,
                                          M, N, K, lda, ldb, ldc,
                                          sAb, sAh, sBb, sBh, sCb, sCh, Hh, fuse);
  };
  auto gemm1 = [&](const _Float16* A, const _Float16* Bw, const float* bias,
                   const float* resid, float* C32, _Float16* C16,
                   int M, int N, int K, int lda, int ldb, int ldc, int fuse) {
    gemmB(A, Bw, bias, resid, C32, C16, M, N, K, lda, ldb, ldc,
          0, 0, 0, 0, 0, 0, 1, 1, fuse);
  };

  // ---- weight conversion ----
  cvt(enc_Wqkv, w_encQKV, 2 * 3 * D_ * D_);
  cvt(enc_Wo, w_encWo, 2 * D_ * D_);
  cvt(enc_Wf1, w_encF1, 2 * DFF_ * D_);
  cvt(enc_Wf2, w_encF2, 2 * D_ * DFF_);
  cvt(dec_Wqkv, w_decQKV, 2 * 3 * D_ * D_);
  cvt(dec_Wo, w_decWo, 2 * D_ * D_);
  cvt(dec_Wf1, w_decF1, DFF_ * D_);
  cvt(dec_Wf2, w_decF2, D_ * DFF_);
  cvt(enc_conv_w, w_encC, D_ * KIM_);
  cvt(dec_conv_w, w_decC, D_ * KIM_);
  cvt(proj_W, w_proj, CIN_ * D_);

  // ---- series stats, normalization, decoder input ----
  k_stats<<<B_ * CIN_, 32, 0, stream>>>(x_enc, mean, stdv);
  k_norm_in<<<(M_ * CIN_ + 255) / 256, 256, 0, stream>>>(x_enc, mean, stdv, xe, M_ * CIN_);
  k_make_xdec<<<(M_ * CIN_ + 255) / 256, 256, 0, stream>>>(xe, xdn, M_ * CIN_);

  // ---- tau projector ----
  k_pconv<<<B_ * CIN_, 32, 0, stream>>>(x_enc, tau_conv, py);
  k_fc<<<(B_ * PH_ + 255) / 256, 256, 0, stream>>>(py, CIN_, stdv, CIN_, tau_W0, tau_b0, ph0, PH_, 1);
  k_fc<<<(B_ * PH_ + 255) / 256, 256, 0, stream>>>(ph0, PH_, nullptr, 0, tau_W1, tau_b1, ph1, PH_, 1);
  k_fc<<<1, 256, 0, stream>>>(ph1, PH_, nullptr, 0, tau_Wout, nullptr, tauv, 1, 2);
  // ---- delta projector ----
  k_pconv<<<B_ * CIN_, 32, 0, stream>>>(x_enc, delta_conv, py);
  k_fc<<<(B_ * PH_ + 255) / 256, 256, 0, stream>>>(py, CIN_, mean, CIN_, delta_W0, delta_b0, ph0, PH_, 1);
  k_fc<<<(B_ * PH_ + 255) / 256, 256, 0, stream>>>(ph0, PH_, nullptr, 0, delta_W1, delta_b1, ph1, PH_, 1);
  k_fc<<<(B_ * S_ + 255) / 256, 256, 0, stream>>>(ph1, PH_, nullptr, 0, delta_Wout, nullptr, deltav, S_, 0);

  const long sSc_b = (long)H_ * S_ * S_, sSc_h = (long)S_ * S_;

  // ---- encoder embedding (circular conv as im2col GEMM) ----
  k_im2col<<<(M_ * KIM_ + 255) / 256, 256, 0, stream>>>(xe, im2c, M_ * KIM_);
  gemm1(im2c, w_encC, nullptr, nullptr, cur, nullptr, M_, D_, KIM_, KIM_, KIM_, D_, 0);
  k_embed_add<<<(M_ * D_ + 255) / 256, 256, 0, stream>>>(cur, cur16, x_mark_enc, temp_W_enc, M_ * D_);

  // ---- encoder layers ----
  for (int l = 0; l < 2; ++l) {
    const _Float16* Wqkv = w_encQKV + (size_t)l * 3 * D_ * D_;
    gemm1(cur16, Wqkv, enc_bqkv + l * 3 * D_, nullptr, nullptr, qkv16,
          M_, 3 * D_, D_, D_, D_, 3 * D_, 1);
    k_vtrans<<<(BH_ * E_ * S_ + 255) / 256, 256, 0, stream>>>(qkv16, vt16, 3 * D_, 2 * D_, BH_ * E_ * S_);
    gemmB(qkv16, qkv16 + D_, nullptr, nullptr, scores, nullptr,
          S_, S_, E_, 3 * D_, 3 * D_, S_,
          (long)S_ * 3 * D_, 64, (long)S_ * 3 * D_, 64, sSc_b, sSc_h, H_, BH_, 0);
    k_softmax<<<BH_ * S_, 32, 0, stream>>>(scores, probs16, tauv, deltav, S_, 0, 0.125f);
    gemmB(probs16, vt16, nullptr, nullptr, nullptr, ctx16,
          S_, E_, S_, S_, S_, D_,
          sSc_b, sSc_h, (long)H_ * E_ * S_, (long)E_ * S_, (long)S_ * D_, 64, H_, BH_, 0);
    gemm1(ctx16, w_encWo + (size_t)l * D_ * D_, enc_bo + l * D_, cur, t1, nullptr,
          M_, D_, D_, D_, D_, D_, 1);
    k_layernorm<<<M_, 32, 0, stream>>>(t1, enc_ln_g + l * 2 * D_, enc_ln_b + l * 2 * D_, t1n, t1n16);
    gemm1(t1n16, w_encF1 + (size_t)l * DFF_ * D_, enc_bf1 + l * DFF_, nullptr, nullptr, ffh16,
          M_, DFF_, D_, D_, D_, DFF_, 2);
    gemm1(ffh16, w_encF2 + (size_t)l * D_ * DFF_, enc_bf2 + l * D_, t1n, t2, nullptr,
          M_, D_, DFF_, DFF_, DFF_, D_, 1);
    k_layernorm<<<M_, 32, 0, stream>>>(t2, enc_ln_g + l * 2 * D_ + D_, enc_ln_b + l * 2 * D_ + D_, cur, cur16);
  }
  k_layernorm<<<M_, 32, 0, stream>>>(cur, enc_norm_g, enc_norm_b, encF, encF16);

  // ---- decoder embedding ----
  k_im2col<<<(M_ * KIM_ + 255) / 256, 256, 0, stream>>>(xdn, im2c, M_ * KIM_);
  gemm1(im2c, w_decC, nullptr, nullptr, dcur, nullptr, M_, D_, KIM_, KIM_, KIM_, D_, 0);
  k_embed_add<<<(M_ * D_ + 255) / 256, 256, 0, stream>>>(dcur, dcur16, x_mark_dec, temp_W_dec, M_ * D_);

  // ---- decoder layer 0: causal self-attention ----
  gemm1(dcur16, w_decQKV, dec_bqkv, nullptr, nullptr, qkv16, M_, 3 * D_, D_, D_, D_, 3 * D_, 1);
  k_vtrans<<<(BH_ * E_ * S_ + 255) / 256, 256, 0, stream>>>(qkv16, vt16, 3 * D_, 2 * D_, BH_ * E_ * S_);
  gemmB(qkv16, qkv16 + D_, nullptr, nullptr, scores, nullptr,
        S_, S_, E_, 3 * D_, 3 * D_, S_,
        (long)S_ * 3 * D_, 64, (long)S_ * 3 * D_, 64, sSc_b, sSc_h, H_, BH_, 0);
  k_softmax<<<BH_ * S_, 32, 0, stream>>>(scores, probs16, tauv, nullptr, S_, 1, 0.125f);
  gemmB(probs16, vt16, nullptr, nullptr, nullptr, ctx16,
        S_, E_, S_, S_, S_, D_,
        sSc_b, sSc_h, (long)H_ * E_ * S_, (long)E_ * S_, (long)S_ * D_, 64, H_, BH_, 0);
  gemm1(ctx16, w_decWo, dec_bo, dcur, t1, nullptr, M_, D_, D_, D_, D_, D_, 1);
  k_layernorm<<<M_, 32, 0, stream>>>(t1, dec_ln_g, dec_ln_b, t1n, t1n16);

  // ---- decoder layer 0: cross-attention vs encoder ----
  gemm1(t1n16, w_decQKV + (size_t)3 * D_ * D_, dec_bqkv + 3 * D_, nullptr, nullptr, qg16,
        M_, D_, D_, D_, D_, D_, 1);
  gemm1(encF16, w_decQKV + (size_t)4 * D_ * D_, dec_bqkv + 4 * D_, nullptr, nullptr, kv16,
        M_, 2 * D_, D_, D_, D_, 2 * D_, 1);
  k_vtrans<<<(BH_ * E_ * S_ + 255) / 256, 256, 0, stream>>>(kv16, vt16, 2 * D_, D_, BH_ * E_ * S_);
  gemmB(qg16, kv16, nullptr, nullptr, scores, nullptr,
        S_, S_, E_, D_, 2 * D_, S_,
        (long)S_ * D_, 64, (long)S_ * 2 * D_, 64, sSc_b, sSc_h, H_, BH_, 0);
  k_softmax<<<BH_ * S_, 32, 0, stream>>>(scores, probs16, tauv, deltav, S_, 0, 0.125f);
  gemmB(probs16, vt16, nullptr, nullptr, nullptr, ctx16,
        S_, E_, S_, S_, S_, D_,
        sSc_b, sSc_h, (long)H_ * E_ * S_, (long)E_ * S_, (long)S_ * D_, 64, H_, BH_, 0);
  gemm1(ctx16, w_decWo + (size_t)D_ * D_, dec_bo + D_, t1n, t2, nullptr, M_, D_, D_, D_, D_, D_, 1);
  k_layernorm<<<M_, 32, 0, stream>>>(t2, dec_ln_g + D_, dec_ln_b + D_, t2n, t2n16);

  // ---- decoder FFN ----
  gemm1(t2n16, w_decF1, dec_bf1, nullptr, nullptr, ffh16, M_, DFF_, D_, D_, D_, DFF_, 2);
  gemm1(ffh16, w_decF2, dec_bf2, t2n, t1, nullptr, M_, D_, DFF_, DFF_, DFF_, D_, 1);
  k_layernorm<<<M_, 32, 0, stream>>>(t1, dec_ln_g + 2 * D_, dec_ln_b + 2 * D_, dcur, dcur16);
  k_layernorm<<<M_, 32, 0, stream>>>(dcur, dec_norm_g, dec_norm_b, dF, dF16);

  // ---- projection + de-normalization ----
  gemm1(dF16, w_proj, proj_b, nullptr, pout, nullptr, M_, CIN_, D_, D_, D_, CIN_, 1);
  k_denorm<<<(M_ * CIN_ + 255) / 256, 256, 0, stream>>>(pout, mean, stdv, (float*)d_out, M_ * CIN_);
}
